// RadioAttention_10900626997293
// MI455X (gfx1250) — matrix operand order repaired and checked by reading, not compiled, not run on hardware
//
#include <hip/hip_runtime.h>
#include <hip/hip_bf16.h>

// Problem dims (fixed by reference): B=16, S=2048, R=512, H2=2048
#define H2      2048
#define RDIM    512
#define MROWS   32768      // B*S
#define M_TILE  32         // rows per block (two 16-row WMMA row-sets)
#define NTHREADS 512       // 16 waves
#define A_STRIDE 520       // 512 + 8 halves pad (bank spread)

typedef _Float16 v8h  __attribute__((ext_vector_type(8)));
typedef _Float16 v16h __attribute__((ext_vector_type(16)));
typedef float    v8f  __attribute__((ext_vector_type(8)));

// ---------------------------------------------------------------------------
// Prepass: W_radio f32 [2048,512] -> f16 in workspace (B operand, reused 1024x)
// ---------------------------------------------------------------------------
__global__ __launch_bounds__(256)
void convert_w_kernel(const float* __restrict__ W, _Float16* __restrict__ Wh, int n)
{
    int i = blockIdx.x * 256 + threadIdx.x;
    if (i < n) Wh[i] = (_Float16)W[i];
}

// ---------------------------------------------------------------------------
// Fused: proj = tanh(radio @ W^T + b); attn = sigmoid((lstm+proj)@Wg^T + bg);
//        out = lstm + attn * proj * gate.
// One block = 32 rows x all 2048 cols. 16 waves: wave(wm,wn) computes rows
// wm*16..+16, cols wn*256..+256 (16 WMMA tiles held in registers).
// ---------------------------------------------------------------------------
__global__ __launch_bounds__(NTHREADS)
void radio_fused_kernel(const float* __restrict__ lstm,
                        const float* __restrict__ radio,
                        const float* __restrict__ gate,
                        const _Float16* __restrict__ Wh,
                        const float* __restrict__ br,
                        const float* __restrict__ Wg,
                        const float* __restrict__ bg,
                        float* __restrict__ out)
{
    __shared__ _Float16 Alds[M_TILE * A_STRIDE];   // radio tile, f16
    __shared__ float attnAcc[M_TILE];
    __shared__ float attnS[M_TILE];
    __shared__ float gateS[M_TILE];

    const int tid  = threadIdx.x;
    const int wave = tid >> 5;
    const int lane = tid & 31;
    const int wm   = wave >> 3;        // 0/1: which 16-row set
    const int wn   = wave & 7;         // 0..7: which 256-col slice
    const int m0   = blockIdx.x * M_TILE;

    // ---- stage radio rows as f16 into LDS (coalesced), init small arrays ----
    #pragma unroll
    for (int i = 0; i < (M_TILE * RDIM) / NTHREADS; ++i) {
        int idx = i * NTHREADS + tid;
        int m = idx >> 9;              // / 512
        int k = idx & (RDIM - 1);
        Alds[m * A_STRIDE + k] = (_Float16)radio[(m0 + m) * RDIM + k];
    }
    if (tid < M_TILE) {
        attnAcc[tid] = 0.0f;
        gateS[tid]   = gate[m0 + tid];
    }
    __syncthreads();

    // ---- GEMM: 16x16x32 f16 WMMA, K = 512 in 16 steps -----------------------
    const int halfsel = lane >> 4;     // 0: lanes 0-15, 1: lanes 16-31
    const int l16     = lane & 15;
    const int col0    = wn * 256;
    const int mbase   = wm * 16;

    v8f acc[16];
    {
        v8f z = {};
        #pragma unroll
        for (int t = 0; t < 16; ++t) acc[t] = z;
    }

    // Single base pointer; per-tile offsets become immediate offsets.
    const _Float16* bbase = Wh + (col0 + l16) * RDIM + halfsel * 16;
    const _Float16* abase = &Alds[(mbase + l16) * A_STRIDE + halfsel * 16];

    for (int k0 = 0; k0 < RDIM; k0 += 32) {
        // A fragment (contiguous, matches B): lanes<16 K[k0..k0+15],
        //                                     lanes>=16 K[k0+16..31]
        v8h a0 = *(const v8h*)(abase + k0);
        v8h a1 = *(const v8h*)(abase + k0 + 8);
        v16h a;
        #pragma unroll
        for (int i = 0; i < 8; ++i) { a[i] = a0[i]; a[i + 8] = a1[i]; }

        // Load ALL B fragments first so the WMMAs overlap the load queue
        // instead of stalling on s_wait_loadcnt 0 per tile.
        v16h bf[16];
        #pragma unroll
        for (int t = 0; t < 16; ++t)
            bf[t] = *(const v16h*)(bbase + k0 + t * (16 * RDIM));

        #pragma unroll
        for (int t = 0; t < 16; ++t)
            acc[t] = __builtin_amdgcn_wmma_f32_16x16x32_f16(
                         false, a, false, bf[t], (short)0, acc[t], false, false);
    }

    // ---- epilogue in registers: bias + tanh => proj ------------------------
    #pragma unroll
    for (int t = 0; t < 16; ++t) {
        const int h = col0 + t * 16 + l16;
        const float brv = br[h];
        #pragma unroll
        for (int r = 0; r < 8; ++r)
            acc[t][r] = tanhf(acc[t][r] + brv);
    }

    // ---- gate dot: sum_h (lstm + proj) * Wg[h] per row ---------------------
    float p[8];
    #pragma unroll
    for (int r = 0; r < 8; ++r) p[r] = 0.0f;

    #pragma unroll
    for (int t = 0; t < 16; ++t) {
        const int h   = col0 + t * 16 + l16;
        const float wgv = Wg[h];
        #pragma unroll
        for (int r = 0; r < 8; ++r) {
            const int m  = mbase + r + 8 * halfsel;   // C-layout row
            const float lv = lstm[(m0 + m) * H2 + h];
            p[r] += (lv + acc[t][r]) * wgv;
        }
    }
    // reduce across the 16 lanes that share a row (wave32, width-16 butterflies)
    #pragma unroll
    for (int r = 0; r < 8; ++r) {
        p[r] += __shfl_xor(p[r], 1, 16);
        p[r] += __shfl_xor(p[r], 2, 16);
        p[r] += __shfl_xor(p[r], 4, 16);
        p[r] += __shfl_xor(p[r], 8, 16);
    }
    if (l16 == 0) {
        #pragma unroll
        for (int r = 0; r < 8; ++r)
            atomicAdd(&attnAcc[mbase + r + 8 * halfsel], p[r]);   // ds_add_f32
    }
    __syncthreads();

    if (tid < M_TILE) {
        float x = attnAcc[tid] + bg[0];
        attnS[tid] = 1.0f / (1.0f + __expf(-x));
    }
    __syncthreads();

    // ---- fused output: out = lstm + attn * proj * gate ---------------------
    float attnR[8], gateR[8];
    #pragma unroll
    for (int r = 0; r < 8; ++r) {
        const int m = mbase + r + 8 * halfsel;
        attnR[r] = attnS[m];
        gateR[r] = gateS[m];
    }
    #pragma unroll
    for (int t = 0; t < 16; ++t) {
        const int h = col0 + t * 16 + l16;
        #pragma unroll
        for (int r = 0; r < 8; ++r) {
            const int m  = mbase + r + 8 * halfsel;
            const int gi = (m0 + m) * H2 + h;
            out[gi] = lstm[gi] + attnR[r] * acc[t][r] * gateR[r];
        }
    }
}

// ---------------------------------------------------------------------------
extern "C" void kernel_launch(void* const* d_in, const int* in_sizes, int n_in,
                              void* d_out, int out_size, void* d_ws, size_t ws_size,
                              hipStream_t stream)
{
    (void)in_sizes; (void)n_in; (void)out_size; (void)ws_size;

    const float* lstm  = (const float*)d_in[0];   // [B,S,2H]
    const float* radio = (const float*)d_in[1];   // [B,S,R]
    const float* gate  = (const float*)d_in[2];   // [B,S]
    const float* Wr    = (const float*)d_in[3];   // [2H,R]
    const float* br    = (const float*)d_in[4];   // [2H]
    const float* Wg    = (const float*)d_in[5];   // [1,2H]
    const float* bg    = (const float*)d_in[6];   // [1]
    float*       out   = (float*)d_out;

    _Float16* Wh = (_Float16*)d_ws;               // 2 MB f16 copy of W_radio

    const int wn = H2 * RDIM;                     // 1,048,576
    convert_w_kernel<<<(wn + 255) / 256, 256, 0, stream>>>(Wr, Wh, wn);

    radio_fused_kernel<<<MROWS / M_TILE, NTHREADS, 0, stream>>>(
        lstm, radio, gate, Wh, br, Wg, bg, out);
}
